// LSTMDecoder_8813272891968
// MI455X (gfx1250) — matrix operand order repaired and checked by reading, not compiled, not run on hardware
//
#include <hip/hip_runtime.h>

// ---------------------------------------------------------------------------
// Problem dims (compile-time, from the reference)
// ---------------------------------------------------------------------------
constexpr int V_ = 10000, E_ = 512, H_ = 1024, B_ = 128, T_ = 64;
constexpr int CTX_ = H_ * 2 * 2;   // 4096
constexpr int G_   = 4 * H_;       // 4096 (gate width)

typedef __attribute__((ext_vector_type(16))) __bf16 v16bf;
typedef __attribute__((ext_vector_type(8)))  __bf16 v8bf;
typedef __attribute__((ext_vector_type(8)))  float  v8f;

#define DEV static __device__ __forceinline__

// ---------------------------------------------------------------------------
// WMMA helpers (CDNA5 V_WMMA_F32_16X16X32_BF16, wave32)
// ---------------------------------------------------------------------------
DEV v8f wmma_bf16(v16bf a, v16bf b, v8f c) {
#if defined(__gfx1250__)
  // 8 args: (neg_a, A, neg_b, B, c_mod, C, reuse_a, reuse_b)
  return __builtin_amdgcn_wmma_f32_16x16x32_bf16(false, a, false, b, (short)0, c,
                                                 false, false);
#else
  return c;  // host pass / other arch: never executed
#endif
}

// A-matrix (16x32, MxK) fragment: elements 0..15 = K[ka..ka+15];
// ka already includes the lane-half shift.
DEV v16bf load_A2(const __bf16* __restrict__ p) {
  v8bf lo = *(const v8bf*)(p);
  v8bf hi = *(const v8bf*)(p + 8);
  return __builtin_shufflevector(lo, hi, 0, 1, 2, 3, 4, 5, 6, 7,
                                 8, 9, 10, 11, 12, 13, 14, 15);
}

// B-matrix (32x16, KxN) fragment: 16 contiguous K values (lane-half shift
// already folded into the pointer).
DEV v16bf load_B2(const __bf16* __restrict__ p) {
  return *(const v16bf*)p;
}

// ---------------------------------------------------------------------------
// One 32x32 (HAS2) or 32x16 output tile per wave; fp32 accumulate.
// All control flow here is scalar (tile indices come from a readfirstlane'd
// wave id), so EXEC stays all-ones across every WMMA.
// Software-pipelined: next k-step fragments are fetched before this step's
// WMMAs issue, so global_load latency overlaps matrix math.
// ---------------------------------------------------------------------------
template <bool HAS2>
DEV void gemm_tile(const __bf16* __restrict__ A0, const __bf16* __restrict__ B0,
                   int K0, const __bf16* __restrict__ A1,
                   const __bf16* __restrict__ B1, int K1,
                   const float* __restrict__ bias,
                   const float* __restrict__ addend, float* __restrict__ C,
                   int N, int mq, int nt0, int lane) {
  const int r  = lane & 15;
  const int hi = (lane >> 4) & 1;
  const int ka = hi << 4;   // A lane-half: +0 / +16 elements
  const int kb = hi << 4;   // B lane-half: +0 / +16 elements

  v8f acc00 = {}, acc10 = {}, acc01 = {}, acc11 = {};

  for (int pass = 0; pass < 2; ++pass) {
    const __bf16* A = pass ? A1 : A0;
    const __bf16* B = pass ? B1 : B0;
    const int K     = pass ? K1 : K0;
    if (A == nullptr) break;

    const __bf16* pa0 = A + (size_t)(mq * 32 + r) * K + ka;
    const __bf16* pa1 = pa0 + (size_t)16 * K;
    const __bf16* pb0 = B + (size_t)(nt0 * 16 + r) * K + kb;
    const __bf16* pb1 = pb0 + (size_t)16 * K;

    v16bf a0 = load_A2(pa0);
    v16bf a1 = load_A2(pa1);
    v16bf b0 = load_B2(pb0);
    v16bf b1 = {};
    if (HAS2) b1 = load_B2(pb1);

    for (int k = 32; k < K; k += 32) {
      // prefetch next k-step while this step's WMMAs execute
      v16bf na0 = load_A2(pa0 + k);
      v16bf na1 = load_A2(pa1 + k);
      v16bf nb0 = load_B2(pb0 + k);
      v16bf nb1 = {};
      if (HAS2) nb1 = load_B2(pb1 + k);

      acc00 = wmma_bf16(a0, b0, acc00);
      acc10 = wmma_bf16(a1, b0, acc10);
      if (HAS2) {
        acc01 = wmma_bf16(a0, b1, acc01);
        acc11 = wmma_bf16(a1, b1, acc11);
      }
      a0 = na0; a1 = na1; b0 = nb0;
      if (HAS2) b1 = nb1;
    }
    acc00 = wmma_bf16(a0, b0, acc00);
    acc10 = wmma_bf16(a1, b0, acc10);
    if (HAS2) {
      acc01 = wmma_bf16(a0, b1, acc01);
      acc11 = wmma_bf16(a1, b1, acc11);
    }
  }

  // D layout: lane 0-15 -> N = nt*16+lane, VGPR i -> M = i / 8+i (lane half)
  const int col0 = nt0 * 16 + r;
  const int col1 = col0 + 16;
  const float bias0 = bias ? bias[col0] : 0.0f;
  const float bias1 = (HAS2 && bias) ? bias[col1] : 0.0f;
  const int row0 = mq * 32 + (hi << 3);
#pragma unroll
  for (int mi = 0; mi < 2; ++mi) {
    const v8f& aN0 = mi ? acc10 : acc00;
    const v8f& aN1 = mi ? acc11 : acc01;
#pragma unroll
    for (int i = 0; i < 8; ++i) {
      const size_t rr = (size_t)(row0 + mi * 16 + i) * N;
      float v0 = aN0[i] + bias0;
      if (addend) v0 += addend[rr + col0];
      C[rr + col0] = v0;
      if (HAS2) {
        float v1 = aN1[i] + bias1;
        if (addend) v1 += addend[rr + col1];
        C[rr + col1] = v1;
      }
    }
  }
}

// Generic NT GEMM: C[M,N] = A0[M,K0]*B0[N,K0]^T (+ A1[M,K1]*B1[N,K1]^T)
//                           (+ bias[N]) (+ addend[M,N]);  M % 32 == 0.
__global__ __launch_bounds__(256)
void gemm_bf16_nt(const __bf16* __restrict__ A0, const __bf16* __restrict__ B0, int K0,
                  const __bf16* __restrict__ A1, const __bf16* __restrict__ B1, int K1,
                  const float* __restrict__ bias, const float* __restrict__ addend,
                  float* __restrict__ C, int M, int N) {
  const int lane = threadIdx.x & 31;
  // Wave id is uniform across the wave; make it explicit so all tile
  // predicates become scalar branches (no EXEC masking around WMMA).
  const int wv = __builtin_amdgcn_readfirstlane(
      (int)((blockIdx.x * blockDim.x + threadIdx.x) >> 5));
  const int Mq    = M >> 5;           // 32-row blocks
  const int Nt    = N >> 4;           // 16-col tiles
  const int Npair = (Nt + 1) >> 1;
  if (wv >= Mq * Npair) return;       // scalar exit
  const int mq  = wv / Npair;
  const int nt0 = (wv % Npair) * 2;
  if (nt0 + 1 < Nt)
    gemm_tile<true>(A0, B0, K0, A1, B1, K1, bias, addend, C, N, mq, nt0, lane);
  else
    gemm_tile<false>(A0, B0, K0, A1, B1, K1, bias, addend, C, N, mq, nt0, lane);
}

// ---------------------------------------------------------------------------
// Elementwise / setup kernels
// ---------------------------------------------------------------------------
__global__ __launch_bounds__(256)
void f32_to_bf16(const float* __restrict__ in, __bf16* __restrict__ out, size_t n) {
  size_t i = (size_t)blockIdx.x * blockDim.x + threadIdx.x;
  size_t stride = (size_t)gridDim.x * blockDim.x;
  for (; i < n; i += stride) out[i] = (__bf16)in[i];
}

// Split W_ih1 (G_, E_+CTX_) into bf16 Wx (G_,E_) and Wh (G_,CTX_)
__global__ __launch_bounds__(256)
void split_w1(const float* __restrict__ w, __bf16* __restrict__ wx,
              __bf16* __restrict__ wh) {
  const size_t total = (size_t)G_ * (E_ + CTX_);
  size_t i = (size_t)blockIdx.x * blockDim.x + threadIdx.x;
  size_t stride = (size_t)gridDim.x * blockDim.x;
  for (; i < total; i += stride) {
    int row = (int)(i / (E_ + CTX_));
    int col = (int)(i % (E_ + CTX_));
    float v = w[i];
    if (col < E_) wx[(size_t)row * E_ + col] = (__bf16)v;
    else          wh[(size_t)row * CTX_ + (col - E_)] = (__bf16)v;
  }
}

__global__ __launch_bounds__(256)
void add_vec(const float* __restrict__ a, const float* __restrict__ b,
             float* __restrict__ o, int n) {
  int i = blockIdx.x * blockDim.x + threadIdx.x;
  if (i < n) o[i] = a[i] + b[i];
}

__global__ __launch_bounds__(256)
void zero_bf16(__bf16* p, size_t n) {
  size_t i = (size_t)blockIdx.x * blockDim.x + threadIdx.x;
  if (i < n) p[i] = (__bf16)0.0f;
}

__global__ __launch_bounds__(256)
void zero_f32(float* p, size_t n) {
  size_t i = (size_t)blockIdx.x * blockDim.x + threadIdx.x;
  if (i < n) p[i] = 0.0f;
}

// Teacher-forced x for step t (dropout-masked embedding of expanded[:,t])
__global__ __launch_bounds__(256)
void make_x_tf(const int* __restrict__ expanded, int t,
               const float* __restrict__ drop, const float* __restrict__ emb,
               __bf16* __restrict__ x) {
  int idx = blockIdx.x * blockDim.x + threadIdx.x;
  if (idx >= B_ * E_) return;
  int b = idx / E_, k = idx % E_;
  int tok = expanded[b * T_ + t];
  x[idx] = (__bf16)(drop[tok] * emb[(size_t)tok * E_ + k]);
}

// x for step s+1: pred embedding (unmasked) where tf_mask==1, else TF input
__global__ __launch_bounds__(256)
void next_x(const int* __restrict__ expanded, const int* __restrict__ tfmask, int s,
            const int* __restrict__ pred, const float* __restrict__ drop,
            const float* __restrict__ emb, __bf16* __restrict__ x) {
  int idx = blockIdx.x * blockDim.x + threadIdx.x;
  if (idx >= B_ * E_) return;
  int b = idx / E_, k = idx % E_;
  if (tfmask[s * B_ + b] == 1) {
    int tok = pred[b];
    x[idx] = (__bf16)emb[(size_t)tok * E_ + k];
  } else {
    int tok = expanded[b * T_ + s + 1];
    x[idx] = (__bf16)(drop[tok] * emb[(size_t)tok * E_ + k]);
  }
}

// LSTM gate nonlinearity: gates (B_,4H) -> update c (f32), emit h (bf16)
__global__ __launch_bounds__(256)
void lstm_cell(const float* __restrict__ gates, float* __restrict__ c,
               __bf16* __restrict__ h) {
  int idx = blockIdx.x * blockDim.x + threadIdx.x;
  if (idx >= B_ * H_) return;
  int b = idx / H_, j = idx % H_;
  const float* g = gates + (size_t)b * G_;
  float gi = g[j], gf = g[H_ + j], gg = g[2 * H_ + j], go = g[3 * H_ + j];
  float si = 1.0f / (1.0f + __expf(-gi));
  float sf = 1.0f / (1.0f + __expf(-gf));
  float so = 1.0f / (1.0f + __expf(-go));
  float cn = sf * c[idx] + si * tanhf(gg);
  c[idx] = cn;
  h[idx] = (__bf16)(so * tanhf(cn));
}

// Row-wise argmax over V_ logits (first-max semantics)
__global__ __launch_bounds__(256)
void argmax_rows(const float* __restrict__ logits, int* __restrict__ out) {
  __shared__ float sv[256];
  __shared__ int   si[256];
  const int b = blockIdx.x;
  const float* row = logits + (size_t)b * V_;
  float best = -__builtin_huge_valf();
  int   bi   = 0x7fffffff;
  for (int j = threadIdx.x; j < V_; j += blockDim.x) {
    float v = row[j];
    if (v > best || (v == best && j < bi)) { best = v; bi = j; }
  }
  sv[threadIdx.x] = best;
  si[threadIdx.x] = bi;
  __syncthreads();
  for (int ofs = 128; ofs > 0; ofs >>= 1) {
    if ((int)threadIdx.x < ofs) {
      float ov = sv[threadIdx.x + ofs];
      int   oi = si[threadIdx.x + ofs];
      if (ov > sv[threadIdx.x] ||
          (ov == sv[threadIdx.x] && oi < si[threadIdx.x])) {
        sv[threadIdx.x] = ov;
        si[threadIdx.x] = oi;
      }
    }
    __syncthreads();
  }
  if (threadIdx.x == 0) out[b] = si[0];
}

// ---------------------------------------------------------------------------
// Host orchestration
// ---------------------------------------------------------------------------
static inline char* wsalloc(char*& p, size_t bytes) {
  char* r = p;
  p += (bytes + 255) & ~(size_t)255;
  return r;
}

extern "C" void kernel_launch(void* const* d_in, const int* in_sizes, int n_in,
                              void* d_out, int out_size, void* d_ws, size_t ws_size,
                              hipStream_t stream) {
  (void)in_sizes; (void)n_in; (void)out_size; (void)ws_size;
  const float* h_n      = (const float*)d_in[0];
  const int*   expanded = (const int*)  d_in[1];
  const int*   tfmask   = (const int*)  d_in[2];
  const float* drop     = (const float*)d_in[3];
  const float* emb      = (const float*)d_in[4];
  const float* W_ih1    = (const float*)d_in[5];
  const float* W_hh1    = (const float*)d_in[6];
  const float* b_ih1    = (const float*)d_in[7];
  const float* b_hh1    = (const float*)d_in[8];
  const float* W_ih2    = (const float*)d_in[9];
  const float* W_hh2    = (const float*)d_in[10];
  const float* b_ih2    = (const float*)d_in[11];
  const float* b_hh2    = (const float*)d_in[12];
  const float* W_out    = (const float*)d_in[13];
  const float* b_out    = (const float*)d_in[14];
  float* out = (float*)d_out;

  // workspace carve-out
  char* p = (char*)d_ws;
  __bf16* w1x  = (__bf16*)wsalloc(p, (size_t)G_ * E_   * 2);  //  4 MB
  __bf16* w1h  = (__bf16*)wsalloc(p, (size_t)G_ * CTX_ * 2);  // 32 MB
  __bf16* whh1 = (__bf16*)wsalloc(p, (size_t)G_ * H_   * 2);  //  8 MB
  __bf16* wih2 = (__bf16*)wsalloc(p, (size_t)G_ * H_   * 2);  //  8 MB
  __bf16* whh2 = (__bf16*)wsalloc(p, (size_t)G_ * H_   * 2);  //  8 MB
  __bf16* wout = (__bf16*)wsalloc(p, (size_t)V_ * H_   * 2);  // 20 MB
  __bf16* hnbf = (__bf16*)wsalloc(p, (size_t)B_ * CTX_ * 2);  //  1 MB
  float*  pre1 = (float*) wsalloc(p, (size_t)B_ * G_   * 4);  //  2 MB
  float*  gat  = (float*) wsalloc(p, (size_t)B_ * G_   * 4);  //  2 MB
  float*  b1s  = (float*) wsalloc(p, (size_t)G_ * 4);
  float*  b2s  = (float*) wsalloc(p, (size_t)G_ * 4);
  __bf16* h1   = (__bf16*)wsalloc(p, (size_t)B_ * H_ * 2);
  __bf16* h2   = (__bf16*)wsalloc(p, (size_t)B_ * H_ * 2);
  float*  c1   = (float*) wsalloc(p, (size_t)B_ * H_ * 4);
  float*  c2   = (float*) wsalloc(p, (size_t)B_ * H_ * 4);
  __bf16* x    = (__bf16*)wsalloc(p, (size_t)B_ * E_ * 2);
  int*    pred = (int*)   wsalloc(p, (size_t)B_ * 4);

  auto gblk = [](size_t n) { return (unsigned)((n + 255) / 256); };

  // one-time conversions (deterministic, redone each launch)
  split_w1<<<gblk((size_t)G_ * (E_ + CTX_)), 256, 0, stream>>>(W_ih1, w1x, w1h);
  f32_to_bf16<<<gblk((size_t)G_ * H_), 256, 0, stream>>>(W_hh1, whh1, (size_t)G_ * H_);
  f32_to_bf16<<<gblk((size_t)G_ * H_), 256, 0, stream>>>(W_ih2, wih2, (size_t)G_ * H_);
  f32_to_bf16<<<gblk((size_t)G_ * H_), 256, 0, stream>>>(W_hh2, whh2, (size_t)G_ * H_);
  f32_to_bf16<<<gblk((size_t)V_ * H_), 256, 0, stream>>>(W_out, wout, (size_t)V_ * H_);
  f32_to_bf16<<<gblk((size_t)B_ * CTX_), 256, 0, stream>>>(h_n, hnbf, (size_t)B_ * CTX_);
  add_vec<<<gblk(G_), 256, 0, stream>>>(b_ih1, b_hh1, b1s, G_);
  add_vec<<<gblk(G_), 256, 0, stream>>>(b_ih2, b_hh2, b2s, G_);
  zero_bf16<<<gblk((size_t)B_ * H_), 256, 0, stream>>>(h1, (size_t)B_ * H_);
  zero_bf16<<<gblk((size_t)B_ * H_), 256, 0, stream>>>(h2, (size_t)B_ * H_);
  zero_f32 <<<gblk((size_t)B_ * H_), 256, 0, stream>>>(c1, (size_t)B_ * H_);
  zero_f32 <<<gblk((size_t)B_ * H_), 256, 0, stream>>>(c2, (size_t)B_ * H_);
  make_x_tf<<<gblk((size_t)B_ * E_), 256, 0, stream>>>(expanded, 0, drop, emb, x);

  auto launch_gemm = [&](const __bf16* A0, const __bf16* B0, int K0,
                         const __bf16* A1, const __bf16* B1, int K1,
                         const float* bias, const float* addend,
                         float* C, int M, int N) {
    int Mq = M / 32, Nt = N / 16, Npair = (Nt + 1) / 2;
    int waves = Mq * Npair;
    int blocks = (waves + 7) / 8;  // 8 wave32 per 256-thread block
    gemm_bf16_nt<<<blocks, 256, 0, stream>>>(A0, B0, K0, A1, B1, K1, bias, addend,
                                             C, M, N);
  };

  // hoisted loop-invariant: pre1 = h_n @ W_ih1[:,E:]^T + b_ih1 + b_hh1
  launch_gemm(hnbf, w1h, CTX_, nullptr, nullptr, 0, b1s, nullptr, pre1, B_, G_);

  for (int s = 0; s < T_ - 1; ++s) {
    // gates1 = x@W1x^T + h1@W_hh1^T + pre1
    launch_gemm(x, w1x, E_, h1, whh1, H_, nullptr, pre1, gat, B_, G_);
    lstm_cell<<<gblk((size_t)B_ * H_), 256, 0, stream>>>(gat, c1, h1);
    // gates2 = h1@W_ih2^T + h2@W_hh2^T + (b_ih2+b_hh2)
    launch_gemm(h1, wih2, H_, h2, whh2, H_, b2s, nullptr, gat, B_, G_);
    lstm_cell<<<gblk((size_t)B_ * H_), 256, 0, stream>>>(gat, c2, h2);
    // logits = h2@W_out^T + b_out  -> d_out[s]
    float* logits = out + (size_t)s * B_ * V_;
    launch_gemm(h2, wout, H_, nullptr, nullptr, 0, b_out, nullptr, logits, B_, V_);
    if (s < T_ - 2) {
      argmax_rows<<<B_, 256, 0, stream>>>(logits, pred);
      next_x<<<gblk((size_t)B_ * E_), 256, 0, stream>>>(expanded, tfmask, s, pred,
                                                        drop, emb, x);
    }
  }
}
